// GraphAttention_76948634075403
// MI455X (gfx1250) — matrix operand order repaired and checked by reading, not compiled, not run on hardware
//
#include <hip/hip_runtime.h>

typedef __attribute__((ext_vector_type(16))) _Float16 v16h;
typedef __attribute__((ext_vector_type(8)))  _Float16 v8h;
typedef __attribute__((ext_vector_type(4)))  _Float16 v4h;
typedef __attribute__((ext_vector_type(8)))  float    v8f;

#define BB 2
#define NN 4096
#define DD 512
#define HH 8
#define DK 64

// ---- CDNA5 WMMA fragment loaders (wave32) -------------------------------
// A-matrix 16x32 f16: lane = row (lane&15); per-lane K chunks at
// kb=(lane>>4)*8: halves [kb..kb+7] in VGPR0-3, [kb+16..kb+23] in VGPR4-7.
__device__ __forceinline__ v16h load_fragA(const _Float16* p, int ld) {
  int lane = threadIdx.x & 31;
  int r = lane & 15;
  int kb = (lane >> 4) << 3;
  const v8h* c0 = (const v8h*)(p + (size_t)r * ld + kb);
  const v8h* c1 = (const v8h*)(p + (size_t)r * ld + kb + 16);
  v8h lo = *c0, hi = *c1;
  v16h a;
#pragma unroll
  for (int i = 0; i < 8; ++i) { a[i] = lo[i]; a[i + 8] = hi[i]; }
  return a;
}

// B-matrix 32x16 f16: lane = column (lane&15); per-lane K chunks at
// kb=(lane>>4)*8: [kb..kb+7] in elems 0-7, [kb+16..kb+23] in elems 8-15.
// p points at column-major storage: column c contiguous with stride ld.
__device__ __forceinline__ v16h load_fragB(const _Float16* p, int ld) {
  int lane = threadIdx.x & 31;
  int c = lane & 15;
  int kb = (lane >> 4) << 3;
  v8h lo = *(const v8h*)(p + (size_t)c * ld + kb);
  v8h hi = *(const v8h*)(p + (size_t)c * ld + kb + 16);
  v16h b;
#pragma unroll
  for (int i = 0; i < 8; ++i) { b[i] = lo[i]; b[i + 8] = hi[i]; }
  return b;
}

__device__ __forceinline__ v8f wmma16(v16h a, v16h b, v8f c) {
  return __builtin_amdgcn_wmma_f32_16x16x32_f16(false, a, false, b, (short)0, c,
                                                false, false);
}

// ---- prep: x -> f16 ------------------------------------------------------
__global__ __launch_bounds__(256) void ga_prep_x(const float* __restrict__ x,
                                                 _Float16* __restrict__ xh) {
  int i = blockIdx.x * 256 + threadIdx.x;
  const float4 v = ((const float4*)x)[i];
  v4h hv;
  hv[0] = (_Float16)v.x; hv[1] = (_Float16)v.y;
  hv[2] = (_Float16)v.z; hv[3] = (_Float16)v.w;
  ((v4h*)xh)[i] = hv;
}

// ---- prep: transpose + convert weights: Wt[n][k] = W[k][n] ---------------
__global__ __launch_bounds__(256) void ga_prep_wt(const float* __restrict__ Wq,
                                                  const float* __restrict__ Wk,
                                                  const float* __restrict__ Wv,
                                                  const float* __restrict__ Wo,
                                                  _Float16* __restrict__ Wts) {
  int z = blockIdx.y;
  const float* W = (z == 0) ? Wq : (z == 1) ? Wk : (z == 2) ? Wv : Wo;
  _Float16* Wt = Wts + (size_t)z * DD * DD;
  int idx = blockIdx.x * 256 + threadIdx.x;
  int k = idx >> 9, n = idx & 511;
  Wt[(size_t)n * DD + k] = (_Float16)W[idx];
}

// ---- QKV projection GEMM: [8192x512] x [512x512] per z in {Q,K,V} --------
__global__ __launch_bounds__(256) void ga_qkv_gemm(
    const _Float16* __restrict__ xh, const _Float16* __restrict__ Wts,
    const float* __restrict__ bq, const float* __restrict__ bk,
    const float* __restrict__ bv, _Float16* __restrict__ Qh,
    _Float16* __restrict__ Kh, _Float16* __restrict__ Vt) {
  int z = blockIdx.z;
  const _Float16* Wt = Wts + (size_t)z * DD * DD;
  const float* bias = (z == 0) ? bq : (z == 1) ? bk : bv;
  int wid = threadIdx.x >> 5;
  int mbase = blockIdx.x * 32 + (wid & 1) * 16;
  int nbase = blockIdx.y * 256 + (wid >> 1) * 64;
  v8f c0 = {}, c1 = {}, c2 = {}, c3 = {};
  for (int k0 = 0; k0 < DD; k0 += 32) {
    v16h a = load_fragA(xh + (size_t)mbase * DD + k0, DD);
    c0 = wmma16(a, load_fragB(Wt + (size_t)(nbase +  0) * DD + k0, DD), c0);
    c1 = wmma16(a, load_fragB(Wt + (size_t)(nbase + 16) * DD + k0, DD), c1);
    c2 = wmma16(a, load_fragB(Wt + (size_t)(nbase + 32) * DD + k0, DD), c2);
    c3 = wmma16(a, load_fragB(Wt + (size_t)(nbase + 48) * DD + k0, DD), c3);
  }
  v8f cc[4] = {c0, c1, c2, c3};
  int lane = threadIdx.x & 31;
  int colr = lane & 15;
  int rowoff = (lane >> 4) << 3;
  int m0 = mbase + rowoff;
  int b = m0 >> 12;
  int n0 = m0 & (NN - 1);
  if (z < 2) {
    _Float16* dst = (z == 0) ? Qh : Kh;
#pragma unroll
    for (int t = 0; t < 4; ++t) {
      int col = nbase + t * 16 + colr;
      float bi = bias[col];
      int h = col >> 6, dk = col & 63;
      size_t base = (((size_t)b * HH + h) * NN + n0) * DK + dk;
#pragma unroll
      for (int r = 0; r < 8; ++r)
        dst[base + (size_t)r * DK] = (_Float16)(cc[t][r] + bi);
    }
  } else {  // V transposed: Vt[b][h][dk][n], 8 tokens contiguous per lane
#pragma unroll
    for (int t = 0; t < 4; ++t) {
      int col = nbase + t * 16 + colr;
      float bi = bias[col];
      int h = col >> 6, dk = col & 63;
      v8h pk;
#pragma unroll
      for (int r = 0; r < 8; ++r) pk[r] = (_Float16)(cc[t][r] + bi);
      *(v8h*)(Vt + (((size_t)b * HH + h) * DK + dk) * NN + n0) = pk;
    }
  }
}

// ---- flash attention: per wave 16 queries, stream keys in tiles of 32 ----
__global__ __launch_bounds__(128) void ga_attn(const _Float16* __restrict__ Qh,
                                               const _Float16* __restrict__ Kh,
                                               const _Float16* __restrict__ Vt,
                                               const float* __restrict__ gm,
                                               _Float16* __restrict__ Oh) {
  __shared__ __align__(32) _Float16 Pbuf[4][16][32];
  int wid = threadIdx.x >> 5;
  int lane = threadIdx.x & 31;
  int colr = lane & 15;
  int rowoff = (lane >> 4) << 3;
  int qbase = blockIdx.x * 64 + wid * 16;
  int h = blockIdx.y, b = blockIdx.z;
  const _Float16* Qp = Qh + ((size_t)b * HH + h) * NN * DK;
  const _Float16* Kp = Kh + ((size_t)b * HH + h) * NN * DK;
  const _Float16* Vp = Vt + ((size_t)b * HH + h) * DK * NN;
  const float* gmp = gm + (size_t)b * NN * NN;

  v16h qlo = load_fragA(Qp + (size_t)qbase * DK, DK);
  v16h qhi = load_fragA(Qp + (size_t)qbase * DK + 32, DK);

  v8f o0 = {}, o1 = {}, o2 = {}, o3 = {};
  float mrow[8], lrow[8];
#pragma unroll
  for (int r = 0; r < 8; ++r) { mrow[r] = -1e30f; lrow[r] = 0.f; }
  const float NEGB = -1e30f;
  const float scale = 0.125f;  // 1/sqrt(DK)

  for (int j0 = 0; j0 < NN; j0 += 32) {
    v8f zz = {};
    v8f s0 = wmma16(qlo, load_fragB(Kp + (size_t)j0 * DK, DK), zz);
    s0 = wmma16(qhi, load_fragB(Kp + (size_t)j0 * DK + 32, DK), s0);
    v8f s1 = wmma16(qlo, load_fragB(Kp + (size_t)(j0 + 16) * DK, DK), zz);
    s1 = wmma16(qhi, load_fragB(Kp + (size_t)(j0 + 16) * DK + 32, DK), s1);

    const float* gq = gmp + (size_t)(qbase + rowoff) * NN + j0 + colr;
    float corr[8], psum[8];
#pragma unroll
    for (int r = 0; r < 8; ++r) {
      float g0 = gq[(size_t)r * NN];
      float g1 = gq[(size_t)r * NN + 16];
      s0[r] = (g0 == 0.f) ? NEGB : s0[r] * scale * g0;
      s1[r] = (g1 == 0.f) ? NEGB : s1[r] * scale * g1;
      float tm = fmaxf(s0[r], s1[r]);
      tm = fmaxf(tm, __shfl_xor(tm, 1, 32));
      tm = fmaxf(tm, __shfl_xor(tm, 2, 32));
      tm = fmaxf(tm, __shfl_xor(tm, 4, 32));
      tm = fmaxf(tm, __shfl_xor(tm, 8, 32));
      float mnew = fmaxf(mrow[r], tm);
      corr[r] = __expf(mrow[r] - mnew);
      mrow[r] = mnew;
      float p0 = __expf(s0[r] - mnew);
      float p1 = __expf(s1[r] - mnew);
      Pbuf[wid][rowoff + r][colr] = (_Float16)p0;
      Pbuf[wid][rowoff + r][16 + colr] = (_Float16)p1;
      float ps = p0 + p1;
      ps += __shfl_xor(ps, 1, 32);
      ps += __shfl_xor(ps, 2, 32);
      ps += __shfl_xor(ps, 4, 32);
      ps += __shfl_xor(ps, 8, 32);
      psum[r] = ps;
    }
#pragma unroll
    for (int r = 0; r < 8; ++r) {
      lrow[r] = lrow[r] * corr[r] + psum[r];
      o0[r] *= corr[r]; o1[r] *= corr[r]; o2[r] *= corr[r]; o3[r] *= corr[r];
    }
    __syncthreads();  // Pbuf stores -> A-fragment loads (dscnt flush)
    v16h pa = load_fragA(&Pbuf[wid][0][0], 32);
    o0 = wmma16(pa, load_fragB(Vp + (size_t)0  * NN + j0, NN), o0);
    o1 = wmma16(pa, load_fragB(Vp + (size_t)16 * NN + j0, NN), o1);
    o2 = wmma16(pa, load_fragB(Vp + (size_t)32 * NN + j0, NN), o2);
    o3 = wmma16(pa, load_fragB(Vp + (size_t)48 * NN + j0, NN), o3);
  }
  v8f oc[4] = {o0, o1, o2, o3};
#pragma unroll
  for (int t = 0; t < 4; ++t) {
#pragma unroll
    for (int r = 0; r < 8; ++r) {
      int q = qbase + rowoff + r;
      int col = h * DK + t * 16 + colr;
      Oh[((size_t)b * NN + q) * DD + col] = (_Float16)(oc[t][r] / lrow[r]);
    }
  }
}

// ---- output projection: out = Oh @ Wo + bo (f32 out) ---------------------
__global__ __launch_bounds__(256) void ga_out_gemm(
    const _Float16* __restrict__ Oh, const _Float16* __restrict__ Wot,
    const float* __restrict__ bo, float* __restrict__ out) {
  int wid = threadIdx.x >> 5;
  int mbase = blockIdx.x * 32 + (wid & 1) * 16;
  int nbase = blockIdx.y * 256 + (wid >> 1) * 64;
  v8f c0 = {}, c1 = {}, c2 = {}, c3 = {};
  for (int k0 = 0; k0 < DD; k0 += 32) {
    v16h a = load_fragA(Oh + (size_t)mbase * DD + k0, DD);
    c0 = wmma16(a, load_fragB(Wot + (size_t)(nbase +  0) * DD + k0, DD), c0);
    c1 = wmma16(a, load_fragB(Wot + (size_t)(nbase + 16) * DD + k0, DD), c1);
    c2 = wmma16(a, load_fragB(Wot + (size_t)(nbase + 32) * DD + k0, DD), c2);
    c3 = wmma16(a, load_fragB(Wot + (size_t)(nbase + 48) * DD + k0, DD), c3);
  }
  v8f cc[4] = {c0, c1, c2, c3};
  int lane = threadIdx.x & 31;
  int colr = lane & 15;
  int rowoff = (lane >> 4) << 3;
#pragma unroll
  for (int t = 0; t < 4; ++t) {
    int col = nbase + t * 16 + colr;
    float bi = bo[col];
#pragma unroll
    for (int r = 0; r < 8; ++r)
      out[(size_t)(mbase + rowoff + r) * DD + col] = cc[t][r] + bi;
  }
}

// ---- launch --------------------------------------------------------------
extern "C" void kernel_launch(void* const* d_in, const int* in_sizes, int n_in,
                              void* d_out, int out_size, void* d_ws,
                              size_t ws_size, hipStream_t stream) {
  const float* x  = (const float*)d_in[0];
  const float* gm = (const float*)d_in[1];
  const float* Wq = (const float*)d_in[2];
  const float* bq = (const float*)d_in[3];
  const float* Wk = (const float*)d_in[4];
  const float* bk = (const float*)d_in[5];
  const float* Wv = (const float*)d_in[6];
  const float* bv = (const float*)d_in[7];
  const float* Wo = (const float*)d_in[8];
  const float* bo = (const float*)d_in[9];
  float* out = (float*)d_out;
  char* ws = (char*)d_ws;

  // workspace layout (bytes)
  _Float16* xh  = (_Float16*)(ws + 0);          // 8 MB  : x in f16
  _Float16* Wts = (_Float16*)(ws + 8388608);    // 2 MB  : 4 transposed weights
  _Float16* Qh  = (_Float16*)(ws + 10485760);   // 8 MB  : [B][H][N][64]
  _Float16* Kh  = (_Float16*)(ws + 18874368);   // 8 MB  : [B][H][N][64]
  _Float16* Vt  = (_Float16*)(ws + 27262976);   // 8 MB  : [B][H][64][N]
  _Float16* Oh  = (_Float16*)(ws + 35651584);   // 8 MB  : [B*N][512]

  ga_prep_x<<<4096, 256, 0, stream>>>(x, xh);
  ga_prep_wt<<<dim3(1024, 4), 256, 0, stream>>>(Wq, Wk, Wv, Wo, Wts);
  ga_qkv_gemm<<<dim3(256, 2, 3), 256, 0, stream>>>(xh, Wts, bq, bk, bv, Qh, Kh,
                                                   Vt);
  ga_attn<<<dim3(64, 8, 2), 128, 0, stream>>>(Qh, Kh, Vt, gm, Oh);
  ga_out_gemm<<<dim3(256, 2), 256, 0, stream>>>(
      Oh, Wts + (size_t)3 * DD * DD, bo, out);
}
